// DMCAMultiHeadAttentionLayer_46583215292563
// MI455X (gfx1250) — compile-verified, matrix operand order checked by reading, not run on hardware
//
#include <hip/hip_runtime.h>
#include <hip/hip_bf16.h>

typedef __attribute__((ext_vector_type(16))) _Float16 v16h;
typedef __attribute__((ext_vector_type(4)))  _Float16 v4h;
typedef __attribute__((ext_vector_type(8)))  float    v8f;

#define D_MODEL 1024
#define N_HEADS 16
#define DH      64
#define BS      256      // attention block size
#define NEGV    (-1.0e9f)

union FragU { v16h h; unsigned u[8]; };

// Fragment-contiguous ("swizzled") LDS tile order: a 16x32 f16 tile = 512
// halves; lane L's 16 fragment halves are contiguous at tile + L*16.
// (row r, k) -> lane = ((k>>3)&1)*16 + (r&15), pos = ((k&16)?8:0) + (k&7).
// Any row-aligned 8-half chunk lands in one contiguous 16-byte slot, so both
// staging (async b128 straight into LDS) and fragment loads (2x ds_load_b128
// at constant offsets) are single-instruction paths.
__device__ __forceinline__ int swz_idx(int r, int k) {
    return (((k >> 3) & 1) * 16 + (r & 15)) * 16 + ((k & 16) ? 8 : 0) + (k & 7);
}

__device__ __forceinline__ v16h load_frag_swz(const _Float16* __restrict__ tile) {
    const int lane = threadIdx.x & 31;
    const uint4* q = (const uint4*)(tile + lane * 16);
    FragU f;
    uint4 lo = q[0];
    uint4 hi = q[1];
    f.u[0] = lo.x; f.u[1] = lo.y; f.u[2] = lo.z; f.u[3] = lo.w;
    f.u[4] = hi.x; f.u[5] = hi.y; f.u[6] = hi.z; f.u[7] = hi.w;
    return f.h;
}

// CDNA5 async copy: 16 bytes global -> LDS per lane, tracked by ASYNCcnt.
// No VGPR data movement; overlaps with WMMA issue.
__device__ __forceinline__ void async_copy16(const _Float16* __restrict__ src,
                                             _Float16* __restrict__ dstLds) {
    unsigned d = (unsigned)(uintptr_t)dstLds;            // LDS byte address
    unsigned long long s = (unsigned long long)(uintptr_t)src;
    asm volatile("global_load_async_to_lds_b128 %0, %1, off"
                 :: "v"(d), "v"(s) : "memory");
}
__device__ __forceinline__ void wait_async() {
    asm volatile("s_wait_asynccnt 0x0" ::: "memory");
}

// ---------------------------------------------------------------------------
// f32 -> f16 conversion (x pre-pass)
// ---------------------------------------------------------------------------
__global__ __launch_bounds__(256)
void cvt_f32_f16(const float* __restrict__ in, _Float16* __restrict__ out, int n4) {
    int i = blockIdx.x * blockDim.x + threadIdx.x;
    int stride = gridDim.x * blockDim.x;
    for (; i < n4; i += stride) {
        float4 v = ((const float4*)in)[i];
        v4h h;
        h.x = (_Float16)v.x; h.y = (_Float16)v.y;
        h.z = (_Float16)v.z; h.w = (_Float16)v.w;
        ((v4h*)out)[i] = h;
    }
}

// ---------------------------------------------------------------------------
// One-time weight transpose+convert: Wt[n][k] (f16) = (f16)W[k][n].
// Removes all f32->f16 work from the GEMM inner loop and gives W the same
// chunk->frag-slot property as A, enabling async staging of both operands.
// ---------------------------------------------------------------------------
__global__ __launch_bounds__(256)
void transpose_cvt_kernel(const float* __restrict__ W, _Float16* __restrict__ Wt,
                          int N) {
    __shared__ float tile[16][17];
    const int tx = threadIdx.x & 15, ty = threadIdx.x >> 4;
    const int k0 = blockIdx.x * 16, n0 = blockIdx.y * 16;
    tile[ty][tx] = W[(size_t)(k0 + ty) * N + n0 + tx];        // coalesced in n
    __syncthreads();
    Wt[(size_t)(n0 + ty) * N + k0 + tx] = (_Float16)tile[tx][ty]; // coalesced in k
}

// ---------------------------------------------------------------------------
// WMMA GEMM: C[M,N] = A[M,K](f16) * Wt[N,K](f16, pre-transposed) + bias
// 128x128 workgroup tile, 8 waves (2x4), 64x32 per wave, BK=32,
// double-buffered with async global->LDS staging of both operands.
// ---------------------------------------------------------------------------
__device__ __forceinline__ void stage_async_tile(const _Float16* __restrict__ G,
                                                 size_t rowStride,
                                                 _Float16* __restrict__ Ls,
                                                 int tid) {
    // 128 rows x 32 halves -> 512 x 16B chunks, 2 async issues per thread.
#pragma unroll
    for (int i = 0; i < 2; ++i) {
        const int idx = tid + i * 256;
        const int r   = idx >> 2;            // row 0..127
        const int cq  = (idx & 3) * 8;       // k chunk: 0,8,16,24
        async_copy16(G + (size_t)r * rowStride + cq,
                     Ls + (r >> 4) * 512 + swz_idx(r, cq));
    }
}

template <bool OUT_F16>
__global__ __launch_bounds__(256)
void wmma_gemm_kernel(const _Float16* __restrict__ A, const _Float16* __restrict__ Wt,
                      const float* __restrict__ bias, void* __restrict__ Cout,
                      int M, int N, int K) {
    __shared__ _Float16 As[2][8 * 512];
    __shared__ _Float16 Bs[2][8 * 512];

    const int tid  = threadIdx.x;
    const int wave = tid >> 5;
    const int m0 = blockIdx.x * 128;
    const int n0 = blockIdx.y * 128;
    const int wmt = (wave >> 2) * 4;      // wave m-tile base (0 or 4)
    const int wnt = (wave & 3) * 2;       // wave n-tile base (0,2,4,6)

    v8f c[4][2] = {};

    stage_async_tile(A  + (size_t)m0 * K, K, As[0], tid);
    stage_async_tile(Wt + (size_t)n0 * K, K, Bs[0], tid);
    wait_async();
    __syncthreads();

    for (int kb = 0; kb < K; kb += 32) {
        const int cur = (kb >> 5) & 1;
        if (kb + 32 < K) {
            stage_async_tile(A  + (size_t)m0 * K + kb + 32, K, As[cur ^ 1], tid);
            stage_async_tile(Wt + (size_t)n0 * K + kb + 32, K, Bs[cur ^ 1], tid);
        }
        v16h a[4], b[2];
#pragma unroll
        for (int i = 0; i < 4; ++i)
            a[i] = load_frag_swz(&As[cur][(wmt + i) * 512]);
#pragma unroll
        for (int j = 0; j < 2; ++j)
            b[j] = load_frag_swz(&Bs[cur][(wnt + j) * 512]);
#pragma unroll
        for (int i = 0; i < 4; ++i)
#pragma unroll
            for (int j = 0; j < 2; ++j)
                c[i][j] = __builtin_amdgcn_wmma_f32_16x16x32_f16(
                    false, a[i], false, b[j], (short)0, c[i][j], false, false);
        wait_async();          // next-buffer copies overlapped the WMMAs above
        __syncthreads();
    }

    // Epilogue: C-frag layout — VGPR r holds row r (lanes 0-15) / row 8+r (16-31)
    const int lane = tid & 31;
    const int colL = lane & 15;
    const int rH   = (lane >> 4) * 8;
#pragma unroll
    for (int i = 0; i < 4; ++i)
#pragma unroll
        for (int j = 0; j < 2; ++j) {
            const int n = n0 + (wnt + j) * 16 + colL;
            const float bv = bias[n];
#pragma unroll
            for (int r = 0; r < 8; ++r) {
                const size_t m = (size_t)(m0 + (wmt + i) * 16 + rH + r);
                const float val = c[i][j][r] + bv;
                if constexpr (OUT_F16)
                    ((_Float16*)Cout)[m * N + n] = (_Float16)val;
                else
                    ((float*)Cout)[m * N + n] = val;
            }
        }
}

// ---------------------------------------------------------------------------
// Block-local causal attention. One workgroup per (block, head, 64-row chunk).
// Dynamic LDS (fragment-swizzled tiles):
//   Qs 4mt x 2ks | Ks 16kt x 2ks | Vt 4dt x 8ks | P 4rt x 8ks  (512 halves each)
//   S 64x256 f32 (row-major scratch for softmax)
// = 168 KB (CDNA5's 320 KB/WGP budget; impossible on CDNA4's 64 KB).
// ---------------------------------------------------------------------------
__global__ __launch_bounds__(256)
void attn_kernel(const _Float16* __restrict__ qh, const _Float16* __restrict__ kh,
                 const _Float16* __restrict__ vh, _Float16* __restrict__ oh) {
    extern __shared__ _Float16 smem[];
    _Float16* Qs = smem;                   //  4096 halves (8 tiles)
    _Float16* Ks = smem + 4096;            // 16384 halves (32 tiles)
    _Float16* Vt = smem + 20480;           // 16384 halves (32 tiles)
    _Float16* P  = smem + 36864;           // 16384 halves (32 tiles)
    float*    S  = (float*)(smem + 53248); // 64*256 f32

    const int head   = blockIdx.y;
    const int blk    = blockIdx.x >> 2;
    const int qchunk = blockIdx.x & 3;
    const int tid    = threadIdx.x;
    const size_t tok0 = (size_t)blk * BS;
    const int hbase  = head * DH;
    const int qbase  = qchunk * 64;

    // Stage Q (64x64) and K (256x64): async 16B chunks straight into frag slots.
    for (int i = tid; i < 64 * 8; i += 256) {
        const int r = i >> 3, cq = (i & 7) * 8;          // cq in 0..56
        const int ks = cq >> 5, kk = cq & 31;
        async_copy16(qh + (tok0 + qbase + r) * D_MODEL + hbase + cq,
                     Qs + ((r >> 4) * 2 + ks) * 512 + swz_idx(r, kk));
    }
    for (int i = tid; i < 256 * 8; i += 256) {
        const int r = i >> 3, cq = (i & 7) * 8;
        const int ks = cq >> 5, kk = cq & 31;
        async_copy16(kh + (tok0 + r) * D_MODEL + hbase + cq,
                     Ks + ((r >> 4) * 2 + ks) * 512 + swz_idx(r, kk));
    }
    // Stage V^T: element (d, key) -> tile (dt = d>>4, ks = key>>5).
    for (int i = tid; i < 64 * 256; i += 256) {
        const int key = i >> 6, d = i & 63;              // coalesced along d
        const int ks = key >> 5, kk = key & 31;
        Vt[((d >> 4) * 8 + ks) * 512 + swz_idx(d, kk)] =
            vh[(tok0 + key) * D_MODEL + hbase + d];
    }
    wait_async();
    __syncthreads();

    // Scores: S = (Q K^T) * 1/sqrt(dh). Wave w: row-tile w>>1, 8 key-tiles.
    const int wave = tid >> 5;
    const int lane = tid & 31;
    const int rt   = wave >> 1;            // 0..3 (16-row tiles)
    const int ktb  = (wave & 1) * 8;
    for (int kt = ktb; kt < ktb + 8; ++kt) {
        v8f c = {};
#pragma unroll
        for (int ks = 0; ks < 2; ++ks) {
            v16h a = load_frag_swz(Qs + (rt * 2 + ks) * 512);
            v16h b = load_frag_swz(Ks + (kt * 2 + ks) * 512);
            c = __builtin_amdgcn_wmma_f32_16x16x32_f16(
                false, a, false, b, (short)0, c, false, false);
        }
        const int col   = kt * 16 + (lane & 15);
        const int rbase = rt * 16 + (lane >> 4) * 8;
#pragma unroll
        for (int r = 0; r < 8; ++r)
            S[(rbase + r) * 256 + col] = c[r] * 0.125f;   // 1/sqrt(64)
    }
    __syncthreads();

    // Causal softmax: 4 lanes per row, 64 cols each, wave32 shfl_xor reductions.
    {
        const int row  = tid >> 2;
        const int seg  = tid & 3;
        const int qpos = qbase + row;      // position within 256-block
        const int c0   = seg * 64;
        float mx = NEGV;
        for (int cc = c0; cc < c0 + 64; ++cc)
            if (cc <= qpos) mx = fmaxf(mx, S[row * 256 + cc]);
        mx = fmaxf(mx, __shfl_xor(mx, 1, 4));
        mx = fmaxf(mx, __shfl_xor(mx, 2, 4));
        float sum = 0.f;
        for (int cc = c0; cc < c0 + 64; ++cc) {
            const float e = (cc <= qpos) ? __expf(S[row * 256 + cc] - mx) : 0.f;
            S[row * 256 + cc] = e;
            sum += e;
        }
        sum += __shfl_xor(sum, 1, 4);
        sum += __shfl_xor(sum, 2, 4);
        const float inv = 1.0f / sum;
        for (int cc = c0; cc < c0 + 64; ++cc) {
            const int ks = cc >> 5, kk = cc & 31;
            P[((row >> 4) * 8 + ks) * 512 + swz_idx(row, kk)] =
                (_Float16)(S[row * 256 + cc] * inv);
        }
    }
    __syncthreads();

    // O = P V : wave w -> row-tile rt, two 16-wide dh tiles.
    {
        const int dtb = (wave & 1) * 2;
#pragma unroll
        for (int j = 0; j < 2; ++j) {
            const int dt = dtb + j;
            v8f c = {};
#pragma unroll
            for (int ks = 0; ks < 8; ++ks) {
                v16h a = load_frag_swz(P + (rt * 8 + ks) * 512);
                v16h b = load_frag_swz(Vt + (dt * 8 + ks) * 512);
                c = __builtin_amdgcn_wmma_f32_16x16x32_f16(
                    false, a, false, b, (short)0, c, false, false);
            }
            const int dcol  = dt * 16 + (lane & 15);
            const int rbase = rt * 16 + (lane >> 4) * 8;
#pragma unroll
            for (int r = 0; r < 8; ++r)
                oh[(tok0 + qbase + rbase + r) * D_MODEL + hbase + dcol] = (_Float16)c[r];
        }
    }
}

// ---------------------------------------------------------------------------
extern "C" void kernel_launch(void* const* d_in, const int* in_sizes, int n_in,
                              void* d_out, int out_size, void* d_ws, size_t ws_size,
                              hipStream_t stream) {
    const float* x  = (const float*)d_in[0];
    const float* Wq = (const float*)d_in[1];
    const float* bq = (const float*)d_in[2];
    const float* Wk = (const float*)d_in[3];
    const float* bk = (const float*)d_in[4];
    const float* Wv = (const float*)d_in[5];
    const float* bv = (const float*)d_in[6];
    const float* Wo = (const float*)d_in[7];
    const float* bo = (const float*)d_in[8];

    const int NTOK = in_sizes[0] / D_MODEL;   // 16384
    const size_t E = (size_t)NTOK * D_MODEL;
    const size_t WE = (size_t)D_MODEL * D_MODEL;

    _Float16* xh  = (_Float16*)d_ws;
    _Float16* qh  = xh + E;
    _Float16* kh  = qh + E;
    _Float16* vh  = kh + E;
    _Float16* oh  = vh + E;
    _Float16* Wqt = oh + E;
    _Float16* Wkt = Wqt + WE;
    _Float16* Wvt = Wkt + WE;
    _Float16* Wot = Wvt + WE;

    const int n4 = (int)(E / 4);
    cvt_f32_f16<<<(n4 + 255) / 256, 256, 0, stream>>>(x, xh, n4);

    dim3 tgrid(D_MODEL / 16, D_MODEL / 16);
    transpose_cvt_kernel<<<tgrid, 256, 0, stream>>>(Wq, Wqt, D_MODEL);
    transpose_cvt_kernel<<<tgrid, 256, 0, stream>>>(Wk, Wkt, D_MODEL);
    transpose_cvt_kernel<<<tgrid, 256, 0, stream>>>(Wv, Wvt, D_MODEL);
    transpose_cvt_kernel<<<tgrid, 256, 0, stream>>>(Wo, Wot, D_MODEL);

    dim3 ggrid(NTOK / 128, D_MODEL / 128);
    wmma_gemm_kernel<true ><<<ggrid, 256, 0, stream>>>(xh, Wqt, bq, qh, NTOK, D_MODEL, D_MODEL);
    wmma_gemm_kernel<true ><<<ggrid, 256, 0, stream>>>(xh, Wkt, bk, kh, NTOK, D_MODEL, D_MODEL);
    wmma_gemm_kernel<true ><<<ggrid, 256, 0, stream>>>(xh, Wvt, bv, vh, NTOK, D_MODEL, D_MODEL);

    const int nblocks = NTOK / BS;            // 64
    const size_t attn_lds = (size_t)53248 * sizeof(_Float16) + 64 * 256 * sizeof(float);
    attn_kernel<<<dim3(nblocks * 4, N_HEADS), 256, attn_lds, stream>>>(qh, kh, vh, oh);

    wmma_gemm_kernel<false><<<ggrid, 256, 0, stream>>>(oh, Wot, bo, d_out, NTOK, D_MODEL, D_MODEL);
}
